// RelationalAttentionRGCN_49563922596252
// MI455X (gfx1250) — matrix operand order repaired and checked by reading, not compiled, not run on hardware
//
#include <hip/hip_runtime.h>
#include <hip/hip_bf16.h>
#include <stdint.h>

// Problem constants (match reference)
#define NE_  200000
#define NR_  16
#define D_   128
#define B_   4096
#define S_   64
#define L_   2
#define H_   4
#define HD_  32
#define KR_  (NR_*D_)   // 2048 : K of the RGCN GEMM
#define MT_  32         // rows per block in the layer GEMM
#define MW_  (MT_/16)   // m-tiles per wave (=2)
#define KC_  128        // K-chunk (halves) staged in LDS per step
#define NCH_ (KR_/KC_)  // 16 chunks

typedef _Float16 f16;
typedef __attribute__((ext_vector_type(16))) _Float16 v16h;
typedef __attribute__((ext_vector_type(8)))  float    v8f;

union Frag16Q { v16h h; uint4 q[2]; };

// ---- WMMA fragment loaders (CDNA5 ISA 7.12.2 layouts, wave32) ----
// A (16x32 f16, row-major source, lda in halves):
//   lane L: M = m0 + (L&15); half = L>>4
//   halves [8*half, 8*half+8) and [16+8*half, 16+8*half+8) of the row
//   -> two naturally aligned 16B blocks -> 2x b128 loads.
__device__ inline v16h load_a_frag(const f16* A, int lda, int m0, int k0, int lane) {
  Frag16Q a;
  int half = lane >> 4;
  const char* base = (const char*)(A + (size_t)(m0 + (lane & 15)) * lda + k0);
  a.q[0] = *(const uint4*)(base + half * 16);
  a.q[1] = *(const uint4*)(base + 32 + half * 16);
  return a.h;
}

// B (32x16 f16). Source stored column-major: element (k,n) at Bcol[n*ldb + k].
//   lane L: N = n0 + (L&15); holds halves [8*half, 8*half+8) and [16+8*half, 16+8*half+8) of the column
//   -> two naturally aligned 16B blocks -> 2x b128 loads.
__device__ inline v16h load_b_frag(const f16* Bcol, int ldb, int n0, int k0, int lane) {
  Frag16Q b;
  const uint4* col =
      (const uint4*)(Bcol + (size_t)(n0 + (lane & 15)) * ldb + k0 + (lane >> 4) * 8);
  b.q[0] = col[0];
  b.q[1] = col[2];
  return b.h;
}

__device__ inline v8f wmma16(v16h a, v16h b, v8f c) {
  // (neg_a, A, neg_b, B, c_mod, C, reuse_a, reuse_b)
  return __builtin_amdgcn_wmma_f32_16x16x32_f16(false, a, false, b, (short)0, c,
                                                false, false);
}

__device__ inline float wave_sum32(float x) {
#pragma unroll
  for (int o = 16; o > 0; o >>= 1) x += __shfl_xor(x, o, 32);
  return x;
}

// ---------------- Stage 1: gather node + relation-bucketed neighbor sum ----
__global__ __launch_bounds__(128) void k_gather(
    const int* __restrict__ didx, const int* __restrict__ adj_e,
    const int* __restrict__ adj_r, const float* __restrict__ ew,
    const float* __restrict__ emb,
    float* __restrict__ node_f32, f16* __restrict__ node_f16,
    f16* __restrict__ pre16) {
  int b = blockIdx.x;
  int t = threadIdx.x;                       // 0..127 == embedding dim
  __shared__ float acc[NR_ * D_];
  int e0 = didx[b];
  float nv = emb[(size_t)e0 * D_ + t];
  node_f32[(size_t)b * D_ + t] = nv;
  node_f16[(size_t)b * D_ + t] = (f16)nv;
#pragma unroll
  for (int r = 0; r < NR_; ++r) acc[r * D_ + t] = 0.f;
  for (int s = 0; s < S_; ++s) {             // thread t owns column t: race-free
    int   e = adj_e[b * S_ + s];
    int   r = adj_r[b * S_ + s];
    float w = ew[b * S_ + s];
    acc[r * D_ + t] += w * emb[(size_t)e * D_ + t];
  }
#pragma unroll
  for (int r = 0; r < NR_; ++r)
    pre16[(size_t)b * KR_ + r * D_ + t] = (f16)acc[r * D_ + t];
}

// ------------- Weight prep: f16 conversion (and transpose for W_rgcn) ------
__global__ void k_wrgcn_to_col16(const float* __restrict__ W, f16* __restrict__ Wcol) {
  // W: [L, KR, D] row-major  ->  Wcol: [L, D(n), KR(k)] column-major B
  size_t i = (size_t)blockIdx.x * blockDim.x + threadIdx.x;
  size_t per = (size_t)KR_ * D_;
  if (i >= (size_t)L_ * per) return;
  size_t l = i / per, rem = i % per;
  size_t e = rem / KR_, k = rem % KR_;
  Wcol[i] = (f16)W[l * per + k * D_ + e];
}

__global__ void k_f32_to_f16(const float* __restrict__ in, f16* __restrict__ out, int n) {
  int i = blockIdx.x * blockDim.x + threadIdx.x;
  if (i < n) out[i] = (f16)in[i];
}

// ------------- Stage 2: fused RGCN layer (2 WMMA GEMMs + ReLU + LN) -------
// Block = 256 threads (8 waves), M-tile = 32 rows, waves each own a 16-col
// N-tile with MW_=2 accumulators. A panel is staged through LDS in K-chunks
// of 128 halves with register double-buffering of the next chunk.
__global__ __launch_bounds__(256) void k_layer(
    const f16* __restrict__ pre16,
    const float* __restrict__ node_f32, const f16* __restrict__ node_f16,
    const f16* __restrict__ Wcol,   // [D][KR]
    const f16* __restrict__ res16,  // res_w rows ARE the K-contig columns of res_w^T
    const float* __restrict__ res_b,
    const f16* __restrict__ proj16, const float* __restrict__ proj_b,
    const float* __restrict__ lng,  const float* __restrict__ lnb,
    float* __restrict__ out_f32, f16* __restrict__ out_f16) {
  int m0   = blockIdx.x * MT_;
  int tid  = threadIdx.x;
  int wave = tid >> 5, lane = tid & 31;
  int n0   = wave * 16;                       // 8 waves cover N=128
  __shared__ f16   sA[MT_ * KC_];             // 32x128 halves: A staging, later h
  __shared__ float lds_v[MT_ * D_];
  __shared__ float lds_mu[MT_], lds_rs[MT_];

  // staging map: b128 index q = 2*tid+j ; position q*8 halves in the chunk,
  // i.e. row r = q>>4, half-offset o = (q&15)*8 within the 128-half row.
  int q0 = 2 * tid, q1 = 2 * tid + 1;
  int r0s = q0 >> 4, o0s = (q0 & 15) * 8;
  int r1s = q1 >> 4, o1s = (q1 & 15) * 8;

  v8f acc[MW_] = {};
  uint4 st0, st1;
  // prologue: fetch chunk 0 of the pre-panel into registers
  st0 = *(const uint4*)(pre16 + (size_t)(m0 + r0s) * KR_ + o0s);
  st1 = *(const uint4*)(pre16 + (size_t)(m0 + r1s) * KR_ + o1s);

  for (int c = 0; c < NCH_; ++c) {
    __syncthreads();                          // everyone done reading sA
    ((uint4*)sA)[q0] = st0;
    ((uint4*)sA)[q1] = st1;
    __syncthreads();                          // chunk visible to all waves
    if (c + 1 < NCH_) {                       // prefetch next chunk into regs
      st0 = *(const uint4*)(pre16 + (size_t)(m0 + r0s) * KR_ + (c + 1) * KC_ + o0s);
      st1 = *(const uint4*)(pre16 + (size_t)(m0 + r1s) * KR_ + (c + 1) * KC_ + o1s);
      if (c + 2 < NCH_)
        __builtin_prefetch(pre16 + (size_t)(m0 + r0s) * KR_ + (c + 2) * KC_ + o0s, 0, 1);
    }
#pragma unroll
    for (int ks = 0; ks < KC_ / 32; ++ks) {
      int k0 = ks * 32;
      v16h b = load_b_frag(Wcol, KR_, n0, c * KC_ + k0, lane);
#pragma unroll
      for (int mt = 0; mt < MW_; ++mt) {
        v16h a = load_a_frag(sA, KC_, mt * 16, k0, lane);
        acc[mt] = wmma16(a, b, acc[mt]);
      }
    }
  }

  // += node @ res_w^T  (K = 128): stage the 32x128 node panel once
  __syncthreads();
  ((uint4*)sA)[q0] = *(const uint4*)(node_f16 + (size_t)(m0 + r0s) * D_ + o0s);
  ((uint4*)sA)[q1] = *(const uint4*)(node_f16 + (size_t)(m0 + r1s) * D_ + o1s);
  __syncthreads();
#pragma unroll
  for (int ks = 0; ks < D_ / 32; ++ks) {
    int k0 = ks * 32;
    v16h b = load_b_frag(res16, D_, n0, k0, lane);
#pragma unroll
    for (int mt = 0; mt < MW_; ++mt) {
      v16h a = load_a_frag(sA, KC_, mt * 16, k0, lane);
      acc[mt] = wmma16(a, b, acc[mt]);
    }
  }

  // epilogue 1: h = relu(node + msgs + res + res_b) -> sA (reused as h panel)
  int n = n0 + (lane & 15);
  int half = lane >> 4;
  float rb = res_b[n];
  __syncthreads();                            // done reading node panel
#pragma unroll
  for (int mt = 0; mt < MW_; ++mt)
#pragma unroll
    for (int j = 0; j < 8; ++j) {             // C layout: VGPR j -> M = j+8*half
      int m = mt * 16 + j + 8 * half;
      float h = node_f32[(size_t)(m0 + m) * D_ + n] + acc[mt][j] + rb;
      sA[m * KC_ + n] = (f16)(h > 0.f ? h : 0.f);
    }
  __syncthreads();

  // h @ proj_w^T  (K = 128, A from LDS h panel)
  v8f acc2[MW_] = {};
#pragma unroll
  for (int ks = 0; ks < D_ / 32; ++ks) {
    int k0 = ks * 32;
    v16h b = load_b_frag(proj16, D_, n0, k0, lane);
#pragma unroll
    for (int mt = 0; mt < MW_; ++mt) {
      v16h a = load_a_frag(sA, KC_, mt * 16, k0, lane);
      acc2[mt] = wmma16(a, b, acc2[mt]);
    }
  }
  float pb = proj_b[n];
#pragma unroll
  for (int mt = 0; mt < MW_; ++mt)
#pragma unroll
    for (int j = 0; j < 8; ++j) {
      int m = mt * 16 + j + 8 * half;
      lds_v[m * D_ + n] = acc2[mt][j] + pb + node_f32[(size_t)(m0 + m) * D_ + n];
    }
  __syncthreads();
  if (tid < MT_) {                            // per-row LN stats
    float s = 0.f, s2 = 0.f;
    for (int x = 0; x < D_; ++x) { float v = lds_v[tid * D_ + x]; s += v; s2 += v * v; }
    float mu = s * (1.f / D_);
    lds_mu[tid] = mu;
    lds_rs[tid] = rsqrtf(s2 * (1.f / D_) - mu * mu + 1e-5f);
  }
  __syncthreads();
#pragma unroll
  for (int j = 0; j < (MT_ * D_) / 256; ++j) {
    int idx = tid * ((MT_ * D_) / 256) + j;
    int m = idx >> 7, c = idx & (D_ - 1);
    float y = (lds_v[idx] - lds_mu[m]) * lds_rs[m] * lng[c] + lnb[c];
    out_f32[(size_t)(m0 + m) * D_ + c] = y;
    out_f16[(size_t)(m0 + m) * D_ + c] = (f16)y;
  }
}

// ------------- Stage 3: qkv projection [L*B,128]x[128,384] -----------------
__global__ __launch_bounds__(256) void k_qkv(
    const f16* __restrict__ x16, const f16* __restrict__ inp16,
    const float* __restrict__ inb, float* __restrict__ qkv) {
  int m0 = blockIdx.x * 16;
  int wave = threadIdx.x >> 5, lane = threadIdx.x & 31;
  v8f acc[3] = {};
  for (int k0 = 0; k0 < D_; k0 += 32) {
    v16h a = load_a_frag(x16, D_, m0, k0, lane);
#pragma unroll
    for (int t = 0; t < 3; ++t) {
      v16h b = load_b_frag(inp16, D_, wave * 16 + t * 128, k0, lane);
      acc[t] = wmma16(a, b, acc[t]);
    }
  }
  int half = lane >> 4;
#pragma unroll
  for (int t = 0; t < 3; ++t) {
    int n = wave * 16 + t * 128 + (lane & 15);
    float bb = inb[n];
#pragma unroll
    for (int j = 0; j < 8; ++j)
      qkv[(size_t)(m0 + j + 8 * half) * (3 * D_) + n] = acc[t][j] + bb;
  }
}

// ------------- Stage 4: 2-token attention; emit mean_l(o_l) ---------------
__global__ __launch_bounds__(128) void k_attn(const float* __restrict__ qkv,
                                              f16* __restrict__ obar16) {
  int b = blockIdx.x;
  int h = threadIdx.x >> 5, lane = threadIdx.x & 31;   // wave per head, lane = hd
  const float scale = 0.17677669529663687f;            // 1/sqrt(32)
  const float* r0 = qkv + (size_t)(0 * B_ + b) * (3 * D_);
  const float* r1 = qkv + (size_t)(1 * B_ + b) * (3 * D_);
  float q0 = r0[h * HD_ + lane],            q1 = r1[h * HD_ + lane];
  float k0 = r0[D_ + h * HD_ + lane],       k1 = r1[D_ + h * HD_ + lane];
  float v0 = r0[2 * D_ + h * HD_ + lane],   v1 = r1[2 * D_ + h * HD_ + lane];
  float s00 = wave_sum32(q0 * k0) * scale, s01 = wave_sum32(q0 * k1) * scale;
  float s10 = wave_sum32(q1 * k0) * scale, s11 = wave_sum32(q1 * k1) * scale;
  float mA = fmaxf(s00, s01), mB = fmaxf(s10, s11);
  float e00 = __expf(s00 - mA), e01 = __expf(s01 - mA);
  float e10 = __expf(s10 - mB), e11 = __expf(s11 - mB);
  float o0 = (e00 * v0 + e01 * v1) / (e00 + e01);
  float o1 = (e10 * v0 + e11 * v1) / (e10 + e11);
  obar16[(size_t)b * D_ + h * HD_ + lane] = (f16)(0.5f * (o0 + o1));
}

// ------------- Stage 5: out_proj + final LayerNorm ------------------------
__global__ __launch_bounds__(256) void k_final(
    const f16* __restrict__ obar16, const f16* __restrict__ outp16,
    const float* __restrict__ outb, const float* __restrict__ g,
    const float* __restrict__ be, float* __restrict__ out) {
  int m0 = blockIdx.x * 16;
  int tid = threadIdx.x, wave = tid >> 5, lane = tid & 31;
  int n0 = wave * 16;
  __shared__ float lds_v[16 * D_];
  __shared__ float lds_mu[16], lds_rs[16];
  v8f acc = {};
  for (int k0 = 0; k0 < D_; k0 += 32) {
    v16h a = load_a_frag(obar16, D_, m0, k0, lane);
    v16h b = load_b_frag(outp16, D_, n0, k0, lane);
    acc = wmma16(a, b, acc);
  }
  int n = n0 + (lane & 15), half = lane >> 4;
  float bb = outb[n];
#pragma unroll
  for (int j = 0; j < 8; ++j) lds_v[(j + 8 * half) * D_ + n] = acc[j] + bb;
  __syncthreads();
  if (tid < 16) {
    float s = 0.f, s2 = 0.f;
    for (int x = 0; x < D_; ++x) { float v = lds_v[tid * D_ + x]; s += v; s2 += v * v; }
    float mu = s * (1.f / D_);
    lds_mu[tid] = mu;
    lds_rs[tid] = rsqrtf(s2 * (1.f / D_) - mu * mu + 1e-5f);
  }
  __syncthreads();
#pragma unroll
  for (int j = 0; j < 8; ++j) {
    int idx = tid * 8 + j;
    int m = idx >> 7, c = idx & (D_ - 1);
    out[(size_t)(m0 + m) * D_ + c] =
        (lds_v[idx] - lds_mu[m]) * lds_rs[m] * g[c] + be[c];
  }
}

// --------------------------------------------------------------------------
extern "C" void kernel_launch(void* const* d_in, const int* in_sizes, int n_in,
                              void* d_out, int out_size, void* d_ws, size_t ws_size,
                              hipStream_t stream) {
  (void)in_sizes; (void)n_in; (void)out_size; (void)ws_size;
  const int*   didx      = (const int*)d_in[0];
  const int*   adj_e     = (const int*)d_in[1];
  const int*   adj_r     = (const int*)d_in[2];
  const float* ew        = (const float*)d_in[3];
  const float* emb       = (const float*)d_in[4];
  const float* W_rgcn    = (const float*)d_in[5];
  const float* res_w     = (const float*)d_in[6];
  const float* res_b     = (const float*)d_in[7];
  const float* proj_w    = (const float*)d_in[8];
  const float* proj_b    = (const float*)d_in[9];
  const float* ln_g      = (const float*)d_in[10];
  const float* ln_b      = (const float*)d_in[11];
  const float* in_proj_w = (const float*)d_in[12];
  const float* in_proj_b = (const float*)d_in[13];
  const float* out_proj_w= (const float*)d_in[14];
  const float* out_proj_b= (const float*)d_in[15];
  const float* fln_g     = (const float*)d_in[16];
  const float* fln_b     = (const float*)d_in[17];

  char* p = (char*)d_ws;
  auto alloc = [&](size_t bytes) {
    char* r = p; p += (bytes + 255) & ~(size_t)255; return r;
  };
  f16*   pre16   = (f16*)  alloc((size_t)B_ * KR_ * 2);        // 16 MB
  float* node0   = (float*)alloc((size_t)B_ * D_ * 4);
  f16*   node016 = (f16*)  alloc((size_t)B_ * D_ * 2);
  float* xf32    = (float*)alloc((size_t)L_ * B_ * D_ * 4);    // layer outputs [L,B,D]
  f16*   xf16    = (f16*)  alloc((size_t)L_ * B_ * D_ * 2);
  f16*   Wcol16  = (f16*)  alloc((size_t)L_ * KR_ * D_ * 2);
  f16*   res16   = (f16*)  alloc((size_t)L_ * D_ * D_ * 2);
  f16*   proj16  = (f16*)  alloc((size_t)L_ * D_ * D_ * 2);
  f16*   inp16   = (f16*)  alloc((size_t)3 * D_ * D_ * 2);
  f16*   outp16  = (f16*)  alloc((size_t)D_ * D_ * 2);
  float* qkv     = (float*)alloc((size_t)L_ * B_ * 3 * D_ * 4);// 12 MB
  f16*   obar16  = (f16*)  alloc((size_t)B_ * D_ * 2);

  // weight prep
  k_wrgcn_to_col16<<<(L_ * KR_ * D_ + 255) / 256, 256, 0, stream>>>(W_rgcn, Wcol16);
  k_f32_to_f16<<<(L_ * D_ * D_ + 255) / 256, 256, 0, stream>>>(res_w,  res16,  L_ * D_ * D_);
  k_f32_to_f16<<<(L_ * D_ * D_ + 255) / 256, 256, 0, stream>>>(proj_w, proj16, L_ * D_ * D_);
  k_f32_to_f16<<<(3 * D_ * D_ + 255) / 256, 256, 0, stream>>>(in_proj_w, inp16, 3 * D_ * D_);
  k_f32_to_f16<<<(D_ * D_ + 255) / 256, 256, 0, stream>>>(out_proj_w, outp16, D_ * D_);

  // gather + relation scatter
  k_gather<<<B_, 128, 0, stream>>>(didx, adj_e, adj_r, ew, emb, node0, node016, pre16);

  // layer 0: node0 -> x[0]
  k_layer<<<B_ / MT_, 256, 0, stream>>>(pre16, node0, node016,
      Wcol16, res16, res_b, proj16, proj_b, ln_g, ln_b, xf32, xf16);
  // layer 1: x[0] -> x[1]
  k_layer<<<B_ / MT_, 256, 0, stream>>>(pre16, xf32, xf16,
      Wcol16 + (size_t)KR_ * D_, res16 + D_ * D_, res_b + D_,
      proj16 + D_ * D_, proj_b + D_, ln_g + D_, ln_b + D_,
      xf32 + (size_t)B_ * D_, xf16 + (size_t)B_ * D_);

  // attention pipeline
  k_qkv<<<(L_ * B_) / 16, 256, 0, stream>>>(xf16, inp16, in_proj_b, qkv);
  k_attn<<<B_, 128, 0, stream>>>(qkv, obar16);
  k_final<<<B_ / 16, 256, 0, stream>>>(obar16, outp16, out_proj_b, fln_g, fln_b,
                                       (float*)d_out);
}
